// MaskRCNN_64312840290401
// MI455X (gfx1250) — matrix operand order repaired and checked by reading, not compiled, not run on hardware
//
#include <hip/hip_runtime.h>
#include <hip/hip_bf16.h>
#include <math.h>

// ---------------- CDNA5 WMMA types ----------------
typedef __attribute__((ext_vector_type(16))) _Float16 v16h;
typedef __attribute__((ext_vector_type(8)))  _Float16 v8h;
typedef __attribute__((ext_vector_type(8)))  float    v8f;

#define FH 64
#define FW 64
#define FC 256
#define CROP_H 7
#define CROP_W 7
#define NMS_T 1024
#define NMS_MAXM 32   // supports N up to 32768 per batch

#define KPAD 224      // K dim 196 padded to 7*32
#define MPAD 64       // M dim 49 padded to 4*16

// =====================================================================
// Kernel 1: greedy NMS, one 1024-thread workgroup per batch.
// Scores + validity bitmask in registers (20 items/thread for N=20480).
// Each serial round: wave32 shuffle argmax -> 32 partials in LDS ->
// wave-0 shuffle reduce -> broadcast winner -> parallel IoU mask update.
// Only 3 workgroup barriers per round on the serial critical path.
// =====================================================================
__global__ __launch_bounds__(NMS_T)
void nms_kernel(const float* __restrict__ scores,
                const float* __restrict__ boxes,
                int N, int R, int M,
                int* __restrict__ sel_ws, int* __restrict__ sel_out)
{
    const int bi = blockIdx.x;
    const int t  = threadIdx.x;
    const int lane = t & 31;
    const int wv   = t >> 5;
    const float*  sc = scores + (long)bi * N;
    const float4* bx = (const float4*)boxes + (long)bi * N;

    float    s[NMS_MAXM];
    unsigned vmask = 0u;
    for (int j = 0; j < M; ++j) {
        int i = j * NMS_T + t;
        if (i < N) { s[j] = sc[i]; vmask |= (1u << j); }
        else       { s[j] = -INFINITY; }
    }

    __shared__ float wredv[32];
    __shared__ int   wredi[32];
    __shared__ float cb[4];
    __shared__ int   hasL;

    for (int round = 0; round < R; ++round) {
        // local argmax (tie -> smaller index, matching jnp.argmax)
        float bestv = -INFINITY; int besti = 0x7fffffff;
        for (int j = 0; j < M; ++j) {
            if (vmask & (1u << j)) {
                float v = s[j];
                int   i = j * NMS_T + t;
                if (v > bestv || (v == bestv && i < besti)) { bestv = v; besti = i; }
            }
        }
        // wave32 shuffle reduction
        for (int off = 16; off > 0; off >>= 1) {
            float ov = __shfl_down(bestv, off, 32);
            int   oi = __shfl_down(besti, off, 32);
            if (ov > bestv || (ov == bestv && oi < besti)) { bestv = ov; besti = oi; }
        }
        if (lane == 0) { wredv[wv] = bestv; wredi[wv] = besti; }
        __syncthreads();
        if (wv == 0) {
            bestv = wredv[lane]; besti = wredi[lane];
            for (int off = 16; off > 0; off >>= 1) {
                float ov = __shfl_down(bestv, off, 32);
                int   oi = __shfl_down(besti, off, 32);
                if (ov > bestv || (ov == bestv && oi < besti)) { bestv = ov; besti = oi; }
            }
            if (lane == 0) {
                int has  = (bestv > -INFINITY) ? 1 : 0;
                int selv = has ? besti : 0;
                sel_ws[bi * R + round]  = selv;
                sel_out[bi * R + round] = selv;
                hasL = has;
                float4 c4 = bx[selv];
                cb[0] = c4.x; cb[1] = c4.y; cb[2] = c4.z; cb[3] = c4.w;
            }
        }
        __syncthreads();
        if (hasL) {
            float cy1 = cb[0], cx1 = cb[1], cy2 = cb[2], cx2 = cb[3];
            float carea = (cy2 - cy1) * (cx2 - cx1);
            for (int j = 0; j < M; ++j) {
                if (vmask & (1u << j)) {
                    int i = j * NMS_T + t;
                    float4 o = bx[i];
                    float yy1 = fmaxf(cy1, o.x), xx1 = fmaxf(cx1, o.y);
                    float yy2 = fminf(cy2, o.z), xx2 = fminf(cx2, o.w);
                    float inter = fmaxf(yy2 - yy1, 0.f) * fmaxf(xx2 - xx1, 0.f);
                    float areab = (o.z - o.x) * (o.w - o.y);
                    float iou   = inter / (carea + areab - inter + 1e-8f);
                    if (iou > 0.5f) vmask &= ~(1u << j);
                }
            }
        }
        __syncthreads();   // protect cb/hasL before next round's writer
    }
}

// =====================================================================
// Kernel 2: crop-and-resize as a per-region matmul on the WMMA pipe.
//   out[49,256] = Wmat[49,196] x G[196,256]
// Wmat (bilinear weights, 4 nnz/row) is materialized ONCE per region
// into LDS as f16 [64 x 224]; G is gathered per 64-channel chunk into
// LDS transposed [64ch x 224k]. With the ISA 16-bit A/B layouts, every
// WMMA fragment is exactly two aligned ds_load_b128 per operand, so the
// steady state is ds_load_b128 x4 -> v_wmma_f32_16x16x32_f16 (x7, K=224).
// 256 threads = 8 waves; each wave owns 2 of the 16 (Mtile,Ntile) pairs.
// =====================================================================
__global__ __launch_bounds__(256)
void crop_wmma_kernel(const float* __restrict__ feat,
                      const float* __restrict__ boxes,
                      const int*   __restrict__ sel,
                      float*       __restrict__ out,
                      int R, int N)
{
    const int blk = blockIdx.x;
    const int bi  = blk / R;
    const int r   = blk % R;
    const int tid = threadIdx.x;
    const int lane = tid & 31;
    const int wv   = tid >> 5;

    __shared__ _Float16 Ws[MPAD * KPAD];  // weights, row-major [p][q], 28KB
    __shared__ _Float16 Gt[64 * KPAD];    // gathered grid, [ch][q], 28KB
    __shared__ int   Yrow[14], Xcol[14];
    __shared__ float wyA[CROP_H], wxA[CROP_W];
    __shared__ int   inry[CROP_H], inrx[CROP_W];
    __shared__ float bxs[4];

    if (tid == 0) {
        int idx = sel[bi * R + r];
        const float* bp = boxes + ((long)bi * N + idx) * 4;
        bxs[0] = bp[0]; bxs[1] = bp[1]; bxs[2] = bp[2]; bxs[3] = bp[3];
    }
    __syncthreads();

    if (tid < CROP_H) {
        float y1 = bxs[0], y2 = bxs[2];
        float ys = y1 * (FH - 1) + (float)tid * ((y2 - y1) * (FH - 1) / (CROP_H - 1));
        float y0 = floorf(ys);
        float wy = ys - y0;
        int y0i = (int)y0;      y0i = y0i < 0 ? 0 : (y0i > FH - 1 ? FH - 1 : y0i);
        int y1i = (int)y0 + 1;  y1i = y1i < 0 ? 0 : (y1i > FH - 1 ? FH - 1 : y1i);
        Yrow[2 * tid] = y0i; Yrow[2 * tid + 1] = y1i;
        wyA[tid]  = wy;
        inry[tid] = (ys >= 0.f && ys <= (float)(FH - 1)) ? 1 : 0;
    }
    if (tid >= 32 && tid < 32 + CROP_W) {
        int j = tid - 32;
        float x1 = bxs[1], x2 = bxs[3];
        float xs = x1 * (FW - 1) + (float)j * ((x2 - x1) * (FW - 1) / (CROP_W - 1));
        float x0 = floorf(xs);
        float wx = xs - x0;
        int x0i = (int)x0;      x0i = x0i < 0 ? 0 : (x0i > FW - 1 ? FW - 1 : x0i);
        int x1i = (int)x0 + 1;  x1i = x1i < 0 ? 0 : (x1i > FW - 1 ? FW - 1 : x1i);
        Xcol[2 * j] = x0i; Xcol[2 * j + 1] = x1i;
        wxA[j]  = wx;
        inrx[j] = (xs >= 0.f && xs <= (float)(FW - 1)) ? 1 : 0;
    }
    __syncthreads();

    // ---- Stage 0: materialize Wmat[64][224] in LDS (once per region) ----
    for (int i = tid; i < MPAD * KPAD; i += 256) {
        int p = i / KPAD, q = i - p * KPAD;
        float wgt = 0.f;
        if (p < 49 && q < 196) {
            int iy = p / 7, ix = p % 7;
            int ry = q / 14, rx = q % 14;
            float wyv = wyA[iy], wxv = wxA[ix];
            float fy = ((ry >> 1) == iy) ? ((ry & 1) ? wyv : 1.f - wyv) : 0.f;
            float fx = ((rx >> 1) == ix) ? ((rx & 1) ? wxv : 1.f - wxv) : 0.f;
            wgt = fy * fx;
        }
        Ws[i] = (_Float16)wgt;
    }

    for (int chunk = 0; chunk < 4; ++chunk) {
        __syncthreads();
        // Gather corner-grid slab -> LDS transposed [ch][q]; zero K-pad.
        // Global side stays channel-coalesced (lane -> consecutive c).
        for (int i = tid; i < KPAD * 64; i += 256) {
            int q = i >> 6, c = i & 63;
            float v = 0.f;
            if (q < 196) {
                int ry = q / 14, rx = q % 14;
                long fidx = (((long)bi * FH + Yrow[ry]) * FW + Xcol[rx]) * FC
                            + (long)chunk * 64 + c;
                v = feat[fidx];
            }
            Gt[c * KPAD + q] = (_Float16)v;
        }
        __syncthreads();

        for (int tt = 0; tt < 2; ++tt) {
            int tile = wv + tt * 8;          // 0..15, uniform per wave
            int mt = tile >> 2, nt = tile & 3;
            int m  = lane & 15;
            int hi = lane >> 4;
            int p  = mt * 16 + m;            // A-matrix row (sample point)
            int n  = nt * 16 + m;            // B-matrix column (channel)

            // A fragment source: Ws[p][kk*32 + hi*8 + {0..7, 16..23}]
            const v8h* Wrow = (const v8h*)(&Ws[p * KPAD]);
            // B fragment source: Gt[n][kk*32 + hi*8 + {0..7, 16..23}]
            const v8h* Grow = (const v8h*)(&Gt[n * KPAD]);

            v8f acc = {};
#pragma unroll
            for (int kk = 0; kk < 7; ++kk) {
                v8h a_lo = Wrow[kk * 4 + hi];           // k = kk*32+hi*8 .. +7
                v8h a_hi = Wrow[kk * 4 + hi + 2];       // k = +16
                v8h b_lo = Grow[kk * 4 + hi];           // k = kk*32+hi*8 .. +7
                v8h b_hi = Grow[kk * 4 + hi + 2];       // k = +16
                v16h a = __builtin_shufflevector(a_lo, a_hi,
                          0,1,2,3,4,5,6,7,8,9,10,11,12,13,14,15);
                v16h b = __builtin_shufflevector(b_lo, b_hi,
                          0,1,2,3,4,5,6,7,8,9,10,11,12,13,14,15);
                acc = __builtin_amdgcn_wmma_f32_16x16x32_f16(
                          false, a, false, b, (short)0, acc, false, false);
            }
            // D layout: VGPR v -> M = v + hi*8 ; N = lane&15
#pragma unroll
            for (int v = 0; v < 8; ++v) {
                int pm = mt * 16 + v + hi * 8;
                if (pm < 49) {
                    int oiy = pm / 7, oix = pm % 7;
                    int c = chunk * 64 + nt * 16 + (lane & 15);
                    float val = (inry[oiy] && inrx[oix]) ? acc[v] : 0.f;
                    // regions layout: (B, CROP_W, R, CROP_H, C)
                    long oidx = ((((long)bi * CROP_W + oix) * R + r) * CROP_H + oiy) * FC + c;
                    out[oidx] = val;
                }
            }
        }
    }
}

// =====================================================================
extern "C" void kernel_launch(void* const* d_in, const int* in_sizes, int n_in,
                              void* d_out, int out_size, void* d_ws, size_t ws_size,
                              hipStream_t stream) {
    const float* feat   = (const float*)d_in[0];
    const float* scores = (const float*)d_in[1];
    const float* boxes  = (const float*)d_in[2];

    int B = in_sizes[0] / (FH * FW * FC);          // 4
    int N = (B > 0) ? in_sizes[1] / B : 0;         // 20480
    // out = B*7*R*7*256 floats + B*R ints  => out_size = B * R * (49*256 + 1)
    int R = (B > 0) ? out_size / (B * (CROP_H * CROP_W * FC + 1)) : 0;   // 300

    int M = (N + NMS_T - 1) / NMS_T;
    if (M > NMS_MAXM) M = NMS_MAXM;

    int*   sel_ws  = (int*)d_ws;
    float* regions = (float*)d_out;
    int*   sel_out = (int*)d_out + (long)B * CROP_W * R * CROP_H * FC;

    nms_kernel<<<B, NMS_T, 0, stream>>>(scores, boxes, N, R, M, sel_ws, sel_out);
    crop_wmma_kernel<<<B * R, 256, 0, stream>>>(feat, boxes, sel_ws, regions, R, N);
}
